// GPTLanguageModel_50019189129566
// MI455X (gfx1250) — compile-verified, matrix operand order checked by reading, not run on hardware
//
#include <hip/hip_runtime.h>
#include <stdint.h>

// ---------------- model dims ----------------
#define E      384
#define NH     6
#define HS     64
#define LAYERS 6
#define VOCAB  50000
#define BATCH  32
#define SEQ    256
#define MROWS  (BATCH*SEQ)   /* 8192 */
#define FF     (4*E)         /* 1536 */

typedef __bf16 bf16_t;
typedef __attribute__((ext_vector_type(16))) __bf16 v16bf;
typedef __attribute__((ext_vector_type(8)))  float  v8f;

// fp32 -> bf16 round-to-nearest-even
__device__ __forceinline__ uint16_t f2bf(float f) {
  union { float f; uint32_t u; } c; c.f = f;
  uint32_t r = 0x7FFFu + ((c.u >> 16) & 1u);
  return (uint16_t)((c.u + r) >> 16);
}

// ---------------- elementwise kernels ----------------
__global__ void k_f32_to_bf16(const float* __restrict__ src,
                              uint16_t* __restrict__ dst, int n) {
  int i = blockIdx.x * blockDim.x + threadIdx.x;
  int stride = gridDim.x * blockDim.x;
  for (; i < n; i += stride) dst[i] = f2bf(src[i]);
}

__global__ void k_embed(const int* __restrict__ idx,
                        const float* __restrict__ tok,
                        const float* __restrict__ pos,
                        float* __restrict__ x) {
  int bt = blockIdx.x;
  int t  = bt & (SEQ - 1);
  int row = idx[bt];
  const float* tr = tok + (size_t)row * E;
  const float* pr = pos + (size_t)t * E;
  float* xr = x + (size_t)bt * E;
  for (int e = threadIdx.x; e < E; e += blockDim.x) xr[e] = tr[e] + pr[e];
}

// one wave32 per row; out = w * x / (||x||/sqrt(E) + eps), emitted as bf16
__global__ void k_rmsnorm(const float* __restrict__ x,
                          const float* __restrict__ w,
                          uint16_t* __restrict__ out, int M) {
  int wave = threadIdx.x >> 5, lane = threadIdx.x & 31;
  int row = blockIdx.x * 8 + wave;
  if (row >= M) return;
  const float* xr = x + (size_t)row * E;
  float ss = 0.f;
#pragma unroll
  for (int e = lane; e < E; e += 32) { float v = xr[e]; ss += v * v; }
#pragma unroll
  for (int off = 16; off >= 1; off >>= 1) ss += __shfl_xor(ss, off, 32);
  // norm = sqrt(ss) * E^-0.5 ; inv = 1/(norm+eps)
  float inv = 1.0f / (sqrtf(ss) * 0.05103103630798287f + 1e-6f);
  uint16_t* orow = out + (size_t)row * E;
#pragma unroll
  for (int e = lane; e < E; e += 32) orow[e] = f2bf(w[e] * xr[e] * inv);
}

__global__ void k_silu_mul(const float* __restrict__ g,
                           const float* __restrict__ u,
                           uint16_t* __restrict__ out, int n) {
  int i = blockIdx.x * blockDim.x + threadIdx.x;
  int stride = gridDim.x * blockDim.x;
  for (; i < n; i += stride) {
    float gv = g[i];
    float s = gv / (1.0f + __expf(-gv));
    out[i] = f2bf(s * u[i]);
  }
}

// ---- WMMA GEMM: out[M,N](f32) = A[M,K](bf16) * W[N,K]^T (+bias)(+res) ----
// Register-blocked 2x2: each wave computes a 32x32 macro-tile (4 f32
// accumulators = 32 VGPRs, 4 fragments = 32 VGPRs -> no spills). Per K-step of
// 32: 2 A-frags + 2 B-frags (8 x b128 loads) feed 4 independent
// v_wmma_f32_16x16x32_bf16 -> 2 loads/WMMA, no D->A/B RAW hazards.
struct alignas(16) BF8 { bf16_t h[8]; };

// A-operand layout (16-bit, 16x32): lanes 0-15 = rows, K 0-7 (v0-3) & 16-23 (v4-7);
// lanes 16-31 = same rows, K 8-15 & 24-31. B-operand loads rows of W (cols of B)
// with the mirrored layout.
__device__ __forceinline__ v16bf load_frag(const bf16_t* __restrict__ p) {
  const BF8* p0 = (const BF8*)p;        // K kh..kh+7   (16B aligned)
  const BF8* p1 = (const BF8*)(p + 16); // K kh+16..kh+23
  v16bf f;
#pragma unroll
  for (int i = 0; i < 8; ++i) { f[i] = p0->h[i]; f[i + 8] = p1->h[i]; }
  return f;
}

__global__ __launch_bounds__(256)
void k_gemm_bf16_wmma(const uint16_t* __restrict__ Ab,
                      const uint16_t* __restrict__ Wb,
                      const float* __restrict__ bias,
                      const float* __restrict__ res,
                      float* __restrict__ out,
                      int N, int K, int ldo) {
  const bf16_t* A = (const bf16_t*)Ab;
  const bf16_t* W = (const bf16_t*)Wb;
  int lane = threadIdx.x & 31;
  int wave = threadIdx.x >> 5;
  int n0 = (blockIdx.x * 8 + wave) * 32;   // wave-uniform
  int m0 = blockIdx.y * 32;
  if (n0 >= N) return;                     // whole wave exits (EXEC stays all-1s)

  // per-lane intra-fragment offsets
  int fr = lane & 15;                      // row within tile
  int fk = (lane & 16) ? 8 : 0;            // K sub-offset for upper lane half

  // wave-uniform clamp of the second (possibly ragged) N tile: loads stay
  // in-bounds (N % 16 == 0); stores are guarded in the epilogue.
  int nr1 = (n0 + 16 < N) ? (n0 + 16) : (N - 16);

  const bf16_t* pA0 = A + (size_t)(m0 + fr) * K + fk;
  const bf16_t* pA1 = A + (size_t)(m0 + 16 + fr) * K + fk;
  const bf16_t* pB0 = W + (size_t)(n0 + fr) * K + fk;
  const bf16_t* pB1 = W + (size_t)(nr1 + fr) * K + fk;

  v8f acc00 = {}, acc01 = {}, acc10 = {}, acc11 = {};

  for (int k0 = 0; k0 < K; k0 += 32) {
    v16bf a0 = load_frag(pA0 + k0);
    v16bf a1 = load_frag(pA1 + k0);
    v16bf b0 = load_frag(pB0 + k0);
    v16bf b1 = load_frag(pB1 + k0);
    acc00 = __builtin_amdgcn_wmma_f32_16x16x32_bf16(false, a0, false, b0, (short)0, acc00, false, false);
    acc01 = __builtin_amdgcn_wmma_f32_16x16x32_bf16(false, a0, false, b1, (short)0, acc01, false, false);
    acc10 = __builtin_amdgcn_wmma_f32_16x16x32_bf16(false, a1, false, b0, (short)0, acc10, false, false);
    acc11 = __builtin_amdgcn_wmma_f32_16x16x32_bf16(false, a1, false, b1, (short)0, acc11, false, false);
  }

  // C/D layout: VGPR r -> lanes 0-15: (M=r, N=lane), lanes 16-31: (M=8+r, N=lane-16)
  int colin = lane & 15;
  int rb    = (lane & 16) ? 8 : 0;
#pragma unroll
  for (int j = 0; j < 2; ++j) {
    int nj = n0 + j * 16;
    if (nj >= N) break;                    // wave-uniform
    int col = nj + colin;
    float badd = bias ? bias[col] : 0.0f;
    const v8f* arow0 = j ? &acc01 : &acc00;
    const v8f* arow1 = j ? &acc11 : &acc10;
#pragma unroll
    for (int r = 0; r < 8; ++r) {
      int m = m0 + rb + r;
      size_t o = (size_t)m * ldo + col;
      float v = (*arow0)[r] + badd;
      if (res) v += res[o];
      out[o] = v;
    }
#pragma unroll
    for (int r = 0; r < 8; ++r) {
      int m = m0 + 16 + rb + r;
      size_t o = (size_t)m * ldo + col;
      float v = (*arow1)[r] + badd;
      if (res) v += res[o];
      out[o] = v;
    }
  }
}

// ---------------- attention: one 64-thread block per (b,h,t) ----------------
__global__ void k_attention(const float* __restrict__ qkv, float* __restrict__ o) {
  __shared__ float sc[SEQ];
  __shared__ float red[64];
  __shared__ float qsh[HS];
  int t = blockIdx.x, h = blockIdx.y, b = blockIdx.z;
  int tid = threadIdx.x;  // 64 threads
  const size_t rowStride = 3 * E;
  const float* qrow = qkv + (size_t)(b * SEQ + t) * rowStride + h * HS;
  qsh[tid] = qrow[tid];
  __syncthreads();

  const float scale = 0.125f;  // HS^-0.5
  for (int s = tid; s <= t; s += 64) {
    const float* krow = qkv + (size_t)(b * SEQ + s) * rowStride + E + h * HS;
    float d = 0.f;
#pragma unroll 8
    for (int e = 0; e < HS; ++e) d += qsh[e] * krow[e];
    sc[s] = d * scale;
  }
  __syncthreads();

  float lm = -3.4e38f;
  for (int s = tid; s <= t; s += 64) lm = fmaxf(lm, sc[s]);
  red[tid] = lm; __syncthreads();
  for (int off = 32; off >= 1; off >>= 1) {
    if (tid < off) red[tid] = fmaxf(red[tid], red[tid + off]);
    __syncthreads();
  }
  float mx = red[0];
  __syncthreads();

  float ls = 0.f;
  for (int s = tid; s <= t; s += 64) { float e = __expf(sc[s] - mx); sc[s] = e; ls += e; }
  red[tid] = ls; __syncthreads();
  for (int off = 32; off >= 1; off >>= 1) {
    if (tid < off) red[tid] += red[tid + off];
    __syncthreads();
  }
  float inv = 1.0f / red[0];
  __syncthreads();

  float acc = 0.f;
  for (int s = 0; s <= t; ++s) {
    const float* vrow = qkv + (size_t)(b * SEQ + s) * rowStride + 2 * E + h * HS;
    acc += sc[s] * vrow[tid];
  }
  o[(size_t)(b * SEQ + t) * E + h * HS + tid] = acc * inv;
}

// ---------------- host ----------------
static inline void gemm(const uint16_t* A, const uint16_t* W, const float* bias,
                        const float* res, float* out, int N, int K, int ldo,
                        hipStream_t s) {
  // block = 8 waves, each wave covers 32 columns -> 256 columns per block
  dim3 g((N + 255) / 256, MROWS / 32);
  k_gemm_bf16_wmma<<<g, dim3(256), 0, s>>>(A, W, bias, res, out, N, K, ldo);
}

static inline void conv(const float* src, uint16_t* dst, int n, hipStream_t s) {
  int blocks = (n + 255) / 256; if (blocks > 4096) blocks = 4096;
  k_f32_to_bf16<<<blocks, 256, 0, s>>>(src, dst, n);
}

extern "C" void kernel_launch(void* const* d_in, const int* in_sizes, int n_in,
                              void* d_out, int out_size, void* d_ws, size_t ws_size,
                              hipStream_t stream) {
  (void)in_sizes; (void)n_in; (void)out_size; (void)ws_size;
  const int*   idx     = (const int*)d_in[0];
  const float* tok_emb = (const float*)d_in[1];
  const float* pos_emb = (const float*)d_in[2];
  const float* wq      = (const float*)d_in[3];
  const float* wk      = (const float*)d_in[4];
  const float* wv      = (const float*)d_in[5];
  const float* w_proj  = (const float*)d_in[6];
  const float* b_proj  = (const float*)d_in[7];
  const float* ln1_w   = (const float*)d_in[8];
  const float* ln2_w   = (const float*)d_in[9];
  const float* w_gate  = (const float*)d_in[10];
  const float* b_gate  = (const float*)d_in[11];
  const float* w_up    = (const float*)d_in[12];
  const float* b_up    = (const float*)d_in[13];
  const float* w_down  = (const float*)d_in[14];
  const float* b_down  = (const float*)d_in[15];
  const float* lnf_w   = (const float*)d_in[16];
  const float* lm_w    = (const float*)d_in[17];
  const float* lm_b    = (const float*)d_in[18];

  char* ws = (char*)d_ws;
  size_t off = 0;
  auto alloc = [&](size_t bytes) -> void* {
    void* p = ws + off;
    off = (off + bytes + 255) & ~(size_t)255;
    return p;
  };

  const size_t SQW = (size_t)LAYERS * E * E;     // 884736
  const size_t SFW = (size_t)LAYERS * FF * E;    // 3538944
  uint16_t* wq_bf   = (uint16_t*)alloc(SQW * 2);
  uint16_t* wk_bf   = (uint16_t*)alloc(SQW * 2);
  uint16_t* wv_bf   = (uint16_t*)alloc(SQW * 2);
  uint16_t* wproj_bf= (uint16_t*)alloc(SQW * 2);
  uint16_t* wg_bf   = (uint16_t*)alloc(SFW * 2);
  uint16_t* wu_bf   = (uint16_t*)alloc(SFW * 2);
  uint16_t* wd_bf   = (uint16_t*)alloc(SFW * 2);
  uint16_t* lmw_bf  = (uint16_t*)alloc((size_t)VOCAB * E * 2);
  float*    x       = (float*)   alloc((size_t)MROWS * E * 4);
  float*    x2      = (float*)   alloc((size_t)MROWS * E * 4);
  uint16_t* hbf     = (uint16_t*)alloc((size_t)MROWS * E * 2);
  float*    qkvbuf  = (float*)   alloc((size_t)MROWS * 3 * E * 4);
  float*    obuf    = (float*)   alloc((size_t)MROWS * E * 4);
  uint16_t* obf     = (uint16_t*)alloc((size_t)MROWS * E * 2);
  float*    gbuf    = (float*)   alloc((size_t)MROWS * FF * 4);
  float*    ubuf    = (float*)   alloc((size_t)MROWS * FF * 4);
  uint16_t* actbf   = (uint16_t*)alloc((size_t)MROWS * FF * 2);

  // weights -> bf16 (all bf16 weights together ~67MB: resident in 192MB L2)
  conv(wq,     wq_bf,    (int)SQW, stream);
  conv(wk,     wk_bf,    (int)SQW, stream);
  conv(wv,     wv_bf,    (int)SQW, stream);
  conv(w_proj, wproj_bf, (int)SQW, stream);
  conv(w_gate, wg_bf,    (int)SFW, stream);
  conv(w_up,   wu_bf,    (int)SFW, stream);
  conv(w_down, wd_bf,    (int)SFW, stream);
  conv(lm_w,   lmw_bf,   VOCAB * E, stream);

  k_embed<<<MROWS, 128, 0, stream>>>(idx, tok_emb, pos_emb, x);

  for (int l = 0; l < LAYERS; ++l) {
    const size_t qo = (size_t)l * E * E;
    const size_t fo = (size_t)l * FF * E;

    k_rmsnorm<<<MROWS / 8, 256, 0, stream>>>(x, ln1_w + (size_t)l * E, hbf, MROWS);

    // q/k/v GEMMs write strided into qkvbuf[M, 3E]
    gemm(hbf, wq_bf + qo, nullptr, nullptr, qkvbuf + 0,     E, E, 3 * E, stream);
    gemm(hbf, wk_bf + qo, nullptr, nullptr, qkvbuf + E,     E, E, 3 * E, stream);
    gemm(hbf, wv_bf + qo, nullptr, nullptr, qkvbuf + 2 * E, E, E, 3 * E, stream);

    k_attention<<<dim3(SEQ, NH, BATCH), 64, 0, stream>>>(qkvbuf, obuf);

    conv(obuf, obf, MROWS * E, stream);
    // x2 = x + o @ Wproj^T + b_proj
    gemm(obf, wproj_bf + qo, b_proj + (size_t)l * E, x, x2, E, E, E, stream);
    { float* tmp = x; x = x2; x2 = tmp; }

    k_rmsnorm<<<MROWS / 8, 256, 0, stream>>>(x, ln2_w + (size_t)l * E, hbf, MROWS);

    gemm(hbf, wg_bf + fo, b_gate + (size_t)l * FF, nullptr, gbuf, FF, E, FF, stream);
    gemm(hbf, wu_bf + fo, b_up   + (size_t)l * FF, nullptr, ubuf, FF, E, FF, stream);
    k_silu_mul<<<4096, 256, 0, stream>>>(gbuf, ubuf, actbf, MROWS * FF);

    // x2 = x + act @ Wdown^T + b_down   (K = 1536)
    gemm(actbf, wd_bf + fo, b_down + (size_t)l * E, x, x2, E, FF, E, stream);
    { float* tmp = x; x = x2; x2 = tmp; }
  }

  k_rmsnorm<<<MROWS / 8, 256, 0, stream>>>(x, lnf_w, hbf, MROWS);

  // logits: store-bandwidth bound (1.64 GB fp32 out); lm_w (38.4MB bf16) is L2-resident
  gemm(hbf, lmw_bf, lm_b, nullptr, (float*)d_out, VOCAB, E, VOCAB, stream);
}
